// EGNNEncoder_63591285785372
// MI455X (gfx1250) — matrix operand order repaired and checked by reading, not compiled, not run on hardware
//
#include <hip/hip_runtime.h>
#include <hip/hip_bf16.h>

typedef __attribute__((ext_vector_type(16))) _Float16 v16h;
typedef __attribute__((ext_vector_type(8)))  _Float16 v8h;
typedef __attribute__((ext_vector_type(8)))  float    v8f;

#define NN 20000
#define NE 320000
#define NG 128
#define WPB 4            // waves per block (wave32)
#define LSTR 264         // LDS row stride in halves (max h=256 + 8 pad)

__device__ __forceinline__ v16h hcombine(v8h lo, v8h hi) {
  v16h r;
#pragma unroll
  for (int i = 0; i < 8; ++i) { r[i] = lo[i]; r[i + 8] = hi[i]; }
  return r;
}

__device__ __forceinline__ v8f wmma16(v16h a, v16h b, v8f c) {
  return __builtin_amdgcn_wmma_f32_16x16x32_f16(false, a, false, b, (short)0, c, false, false);
}

// ---------- weight packing: f32 [Kact, N] -> f16 B-fragment-contiguous ----------
// layout: P[(((kc*NT + nt)*32 + lane)*16 + j]  where lane<16 holds K=kc*32+(j<8 ? j : 8+j) (col nt*16+lane),
// lane>=16 holds K=kc*32+8+(j<8 ? j : 8+j). Rows >= Kact are zero padding.
__global__ void k_pack_b(const float* __restrict__ W, _Float16* __restrict__ P,
                         int Kact, int Kpad, int N) {
  int idx = blockIdx.x * blockDim.x + threadIdx.x;
  int total = Kpad * N;
  if (idx >= total) return;
  int j    = idx & 15;
  int lane = (idx >> 4) & 31;
  int t    = idx >> 9;
  int NT = N >> 4;
  int nt = t % NT;
  int kc = t / NT;
  int k = kc * 32 + ((lane >> 4) << 3) + (j & 7) + ((j >> 3) << 4);
  int n = nt * 16 + (lane & 15);
  float v = (k < Kact) ? W[(size_t)k * N + n] : 0.0f;
  P[idx] = (_Float16)v;
}

// ---------- small utility kernels ----------
__global__ void k_f32_to_f16(const float* __restrict__ s, _Float16* __restrict__ d, int n) {
  int i = blockIdx.x * blockDim.x + threadIdx.x;
  if (i < n) d[i] = (_Float16)s[i];
}
__global__ void k_copy_f32(float* __restrict__ d, const float* __restrict__ s, int n) {
  int i = blockIdx.x * blockDim.x + threadIdx.x;
  if (i < n) d[i] = s[i];
}
__global__ void k_zero_f32(float* __restrict__ d, int n) {
  int i = blockIdx.x * blockDim.x + threadIdx.x;
  if (i < n) d[i] = 0.0f;
}

__global__ void k_radial(const float* __restrict__ coords, const int* __restrict__ src,
                         const int* __restrict__ dst, _Float16* __restrict__ radial16,
                         float* __restrict__ normed, int E) {
  int e = blockIdx.x * blockDim.x + threadIdx.x;
  if (e >= E) return;
  int s = src[e], d = dst[e];
  float dx = coords[s * 3 + 0] - coords[d * 3 + 0];
  float dy = coords[s * 3 + 1] - coords[d * 3 + 1];
  float dz = coords[s * 3 + 2] - coords[d * 3 + 2];
  float r = dx * dx + dy * dy + dz * dz;
  radial16[e] = (_Float16)r;
  float inv = 1.0f / (sqrtf(r) + 1.0f);
  normed[(size_t)e * 3 + 0] = dx * inv;
  normed[(size_t)e * 3 + 1] = dy * inv;
  normed[(size_t)e * 3 + 2] = dz * inv;
}

// ---------- fused edge MLP: relu(edge_in @ ew1 + b1) @ ew2 + b2 ----------
// edge_in (f16, K padded to 288) is assembled in registers:
//   k 0..127   = nf16[src], k 128..255 = nf16[dst], k 256..271 = eattr16, k 272 = radial16
__global__ void __launch_bounds__(128) k_edge_mlp(
    const _Float16* __restrict__ nf16, const _Float16* __restrict__ eattr16,
    const _Float16* __restrict__ radial16,
    const int* __restrict__ src, const int* __restrict__ dst,
    const _Float16* __restrict__ Pw1, const float* __restrict__ b1,
    const _Float16* __restrict__ Pw2, const float* __restrict__ b2,
    float* __restrict__ edge_out, int Etiles, int h)
{
  __shared__ _Float16 lds[WPB * 16 * LSTR];
  int wave = threadIdx.x >> 5;
  int lane = threadIdx.x & 31;
  int tile = blockIdx.x * WPB + wave;
  bool active = (tile < Etiles);
  int row = lane & 15;
  int hi  = lane >> 4;
  int lo  = hi * 8;
  _Float16* lw = lds + wave * 16 * LSTR;
  int NT = h >> 4;

  if (active) {
    int e = tile * 16 + row;
    int s = src[e], d = dst[e];
    v16h a[9];
#pragma unroll
    for (int c = 0; c < 4; ++c) {
      const _Float16* p = nf16 + (size_t)s * 128 + c * 32 + lo;
      a[c] = hcombine(*(const v8h*)p, *(const v8h*)(p + 16));
    }
#pragma unroll
    for (int c = 0; c < 4; ++c) {
      const _Float16* p = nf16 + (size_t)d * 128 + c * 32 + lo;
      a[4 + c] = hcombine(*(const v8h*)p, *(const v8h*)(p + 16));
    }
    {
      const _Float16* p = eattr16 + (size_t)e * 16 + lo;
      v8h g0 = *(const v8h*)p;
      v16h t;
#pragma unroll
      for (int i = 0; i < 8; ++i) { t[i] = g0[i]; t[i + 8] = (_Float16)0.0f; }
      if (hi == 0) t[8] = radial16[e];   // k==272 lives in lanes 0..15, j==8
      a[8] = t;
    }
    for (int nt = 0; nt < NT; ++nt) {
      v8f acc = {};
#pragma unroll
      for (int kc = 0; kc < 9; ++kc) {
        v16h b = *(const v16h*)(Pw1 + (((size_t)kc * NT + nt) * 32 + lane) * 16);
        acc = wmma16(a[kc], b, acc);
      }
      float bv = b1[nt * 16 + row];
#pragma unroll
      for (int r = 0; r < 8; ++r) {
        float v = acc[r] + bv;
        v = v > 0.0f ? v : 0.0f;
        lw[(hi * 8 + r) * LSTR + nt * 16 + row] = (_Float16)v;
      }
    }
  }
  __syncthreads();
  if (active) {
    int KC2 = h >> 5;
    v8f acc = {};
    for (int kc = 0; kc < KC2; ++kc) {
      const _Float16* p = lw + row * LSTR + kc * 32 + lo;
      v16h a2 = hcombine(*(const v8h*)p, *(const v8h*)(p + 16));
      v16h b = *(const v16h*)(Pw2 + ((size_t)kc * 32 + lane) * 16);
      acc = wmma16(a2, b, acc);
    }
    float bv = b2[row];
    int e0 = tile * 16;
#pragma unroll
    for (int r = 0; r < 8; ++r)
      edge_out[(size_t)(e0 + hi * 8 + r) * 16 + row] = acc[r] + bv;
  }
}

// ---------- coord eps + scatter (segment sums) ----------
__global__ void k_coord_agg(const float* __restrict__ edge_out,
                            const float* __restrict__ normed,
                            const int* __restrict__ src,
                            const float* __restrict__ cw1, const float* __restrict__ cb1,
                            const float* __restrict__ cw2,
                            float* __restrict__ coords2, float* __restrict__ agg,
                            _Float16* __restrict__ eattr16_next, int E)
{
  int e = blockIdx.x * blockDim.x + threadIdx.x;
  if (e >= E) return;
  float eo[16];
  float t0 = cb1[0], t1 = cb1[1];
#pragma unroll
  for (int c = 0; c < 16; ++c) {
    float v = edge_out[(size_t)e * 16 + c];
    eo[c] = v;
    eattr16_next[(size_t)e * 16 + c] = (_Float16)v;
    t0 += v * cw1[c * 2 + 0];
    t1 += v * cw1[c * 2 + 1];
  }
  t0 = t0 > 0.0f ? t0 : 0.0f;
  t1 = t1 > 0.0f ? t1 : 0.0f;
  float eps = t0 * cw2[0] + t1 * cw2[1];
  int s = src[e];
#pragma unroll
  for (int k = 0; k < 3; ++k)
    atomicAdd(&coords2[s * 3 + k], normed[(size_t)e * 3 + k] * eps);
#pragma unroll
  for (int c = 0; c < 16; ++c)
    atomicAdd(&agg[s * 16 + c], eo[c]);
}

// ---------- coord_out = coords2 @ cmw + cmb ; build node_extra f16 [N,32] ----------
__global__ void k_node_extra(const float* __restrict__ coords2,
                             const float* __restrict__ cmw, const float* __restrict__ cmb,
                             const float* __restrict__ agg,
                             float* __restrict__ coords_next,
                             _Float16* __restrict__ extra, int N)
{
  int n = blockIdx.x * blockDim.x + threadIdx.x;
  if (n >= N) return;
  float x = coords2[n * 3 + 0], y = coords2[n * 3 + 1], z = coords2[n * 3 + 2];
#pragma unroll
  for (int j = 0; j < 3; ++j) {
    float v = x * cmw[0 * 3 + j] + y * cmw[1 * 3 + j] + z * cmw[2 * 3 + j] + cmb[j];
    coords_next[n * 3 + j] = v;
    extra[(size_t)n * 32 + 16 + j] = (_Float16)v;
  }
#pragma unroll
  for (int c = 0; c < 16; ++c) extra[(size_t)n * 32 + c] = (_Float16)agg[n * 16 + c];
#pragma unroll
  for (int j = 19; j < 32; ++j) extra[(size_t)n * 32 + j] = (_Float16)0.0f;
}

// ---------- fused node MLP: relu(node_in @ nw1 + b1) @ nw2 + b2 ----------
// node_in (K padded to 160): k 0..127 = nf16, k 128..159 = node_extra
__global__ void __launch_bounds__(128) k_node_mlp(
    const _Float16* __restrict__ nf16, const _Float16* __restrict__ extra,
    const _Float16* __restrict__ Pw1, const float* __restrict__ b1,
    const _Float16* __restrict__ Pw2, const float* __restrict__ b2,
    float* __restrict__ nf_out, int Ntiles, int h)
{
  __shared__ _Float16 lds[WPB * 16 * LSTR];
  int wave = threadIdx.x >> 5;
  int lane = threadIdx.x & 31;
  int tile = blockIdx.x * WPB + wave;
  bool active = (tile < Ntiles);
  int row = lane & 15;
  int hi  = lane >> 4;
  int lo  = hi * 8;
  _Float16* lw = lds + wave * 16 * LSTR;
  int NT = h >> 4;

  if (active) {
    int n = tile * 16 + row;
    v16h a[5];
#pragma unroll
    for (int c = 0; c < 4; ++c) {
      const _Float16* p = nf16 + (size_t)n * 128 + c * 32 + lo;
      a[c] = hcombine(*(const v8h*)p, *(const v8h*)(p + 16));
    }
    {
      const _Float16* p = extra + (size_t)n * 32 + lo;
      a[4] = hcombine(*(const v8h*)p, *(const v8h*)(p + 16));
    }
    for (int nt = 0; nt < NT; ++nt) {
      v8f acc = {};
#pragma unroll
      for (int kc = 0; kc < 5; ++kc) {
        v16h b = *(const v16h*)(Pw1 + (((size_t)kc * NT + nt) * 32 + lane) * 16);
        acc = wmma16(a[kc], b, acc);
      }
      float bv = b1[nt * 16 + row];
#pragma unroll
      for (int r = 0; r < 8; ++r) {
        float v = acc[r] + bv;
        v = v > 0.0f ? v : 0.0f;
        lw[(hi * 8 + r) * LSTR + nt * 16 + row] = (_Float16)v;
      }
    }
  }
  __syncthreads();
  if (active) {
    int KC2 = h >> 5;
    int n0 = tile * 16;
    for (int nt2 = 0; nt2 < 8; ++nt2) {           // OUT = 128 -> 8 N-tiles
      v8f acc = {};
      for (int kc = 0; kc < KC2; ++kc) {
        const _Float16* p = lw + row * LSTR + kc * 32 + lo;
        v16h a2 = hcombine(*(const v8h*)p, *(const v8h*)(p + 16));
        v16h b = *(const v16h*)(Pw2 + (((size_t)kc * 8 + nt2) * 32 + lane) * 16);
        acc = wmma16(a2, b, acc);
      }
      float bv = b2[nt2 * 16 + row];
#pragma unroll
      for (int r = 0; r < 8; ++r)
        nf_out[(size_t)(n0 + hi * 8 + r) * 128 + nt2 * 16 + row] = acc[r] + bv;
    }
  }
}

// ---------- post linear: node_embs = nf16 @ post_w + post_b ----------
__global__ void __launch_bounds__(128) k_post(
    const _Float16* __restrict__ nf16, const _Float16* __restrict__ Pw,
    const float* __restrict__ bias, float* __restrict__ out, int Mtiles)
{
  int wave = threadIdx.x >> 5;
  int lane = threadIdx.x & 31;
  int tile = blockIdx.x * WPB + wave;
  if (tile >= Mtiles * 8) return;
  int mt = tile >> 3, nt = tile & 7;
  int row = lane & 15;
  int hi  = lane >> 4;
  int lo  = hi * 8;
  int n = mt * 16 + row;
  v8f acc = {};
#pragma unroll
  for (int kc = 0; kc < 4; ++kc) {
    const _Float16* p = nf16 + (size_t)n * 128 + kc * 32 + lo;
    v16h a = hcombine(*(const v8h*)p, *(const v8h*)(p + 16));
    v16h b = *(const v16h*)(Pw + (((size_t)kc * 8 + nt) * 32 + lane) * 16);
    acc = wmma16(a, b, acc);
  }
  float bv = bias[nt * 16 + row];
#pragma unroll
  for (int r = 0; r < 8; ++r)
    out[(size_t)(mt * 16 + hi * 8 + r) * 128 + nt * 16 + row] = acc[r] + bv;
}

// ---------- graph pooling ----------
__global__ void k_pool_sum(const float* __restrict__ embs, const int* __restrict__ batch,
                           float* __restrict__ sums, int N) {
  int i = blockIdx.x * blockDim.x + threadIdx.x;
  if (i >= N * 128) return;
  int n = i >> 7, c = i & 127;
  atomicAdd(&sums[batch[n] * 128 + c], embs[i]);
}
__global__ void k_pool_cnt(const int* __restrict__ batch, float* __restrict__ counts, int N) {
  int i = blockIdx.x * blockDim.x + threadIdx.x;
  if (i < N) atomicAdd(&counts[batch[i]], 1.0f);
}
__global__ void k_pool_fin(const float* __restrict__ sums, const float* __restrict__ counts,
                           float* __restrict__ out) {
  int i = blockIdx.x * blockDim.x + threadIdx.x;
  if (i >= NG * 128) return;
  float c = counts[i >> 7];
  out[i] = sums[i] / (c > 1.0f ? c : 1.0f);
}

static inline int cdiv(int a, int b) { return (a + b - 1) / b; }

extern "C" void kernel_launch(void* const* d_in, const int* in_sizes, int n_in,
                              void* d_out, int out_size, void* d_ws, size_t ws_size,
                              hipStream_t stream) {
  (void)in_sizes; (void)n_in; (void)out_size; (void)ws_size;
  const int N = NN, E = NE;
  const float* in_nf     = (const float*)d_in[0];
  const float* in_coords = (const float*)d_in[1];
  const float* in_eattr  = (const float*)d_in[2];
  const int*   src       = (const int*)d_in[3];
  const int*   dst       = src + E;
  const int*   batch     = (const int*)d_in[4];
  // params flattened: per layer (sorted keys): cb1 cmb cmw cw1 cw2 eb1 eb2 ew1 ew2 nb1 nb2 nw1 nw2
  const float* post_b = (const float*)d_in[5 + 4 * 13 + 0];
  const float* post_w = (const float*)d_in[5 + 4 * 13 + 1];

  char* wsb = (char*)d_ws;
  size_t off = 0;
  auto alloc = [&](size_t bytes) -> char* {
    char* p = wsb + off;
    off = (off + bytes + 255) & ~(size_t)255;
    return p;
  };
  float*     nfA      = (float*)alloc((size_t)N * 128 * 4);
  float*     nfB      = (float*)alloc((size_t)N * 128 * 4);
  _Float16*  nf16     = (_Float16*)alloc((size_t)N * 128 * 2);
  _Float16*  eattr16  = (_Float16*)alloc((size_t)E * 16 * 2);
  _Float16*  radial16 = (_Float16*)alloc((size_t)E * 2);
  float*     normed   = (float*)alloc((size_t)E * 3 * 4);
  float*     edge_out = (float*)alloc((size_t)E * 16 * 4);
  float*     cA       = (float*)alloc((size_t)N * 3 * 4);
  float*     cB       = (float*)alloc((size_t)N * 3 * 4);
  float*     coords2  = (float*)alloc((size_t)N * 3 * 4);
  float*     agg      = (float*)alloc((size_t)N * 16 * 4);
  _Float16*  extra    = (_Float16*)alloc((size_t)N * 32 * 2);
  float*     sums     = (float*)alloc((size_t)NG * 128 * 4);
  float*     counts   = (float*)alloc((size_t)NG * 4);
  _Float16*  pe1[4]; _Float16* pe2[4]; _Float16* pn1[4]; _Float16* pn2[4];
  for (int l = 0; l < 4; ++l) {
    int h = (l == 0) ? 256 : 128;
    pe1[l] = (_Float16*)alloc((size_t)288 * h * 2);
    pe2[l] = (_Float16*)alloc((size_t)h * 16 * 2);
    pn1[l] = (_Float16*)alloc((size_t)160 * h * 2);
    pn2[l] = (_Float16*)alloc((size_t)h * 128 * 2);
  }
  _Float16* ppost = (_Float16*)alloc((size_t)128 * 128 * 2);

  // ---- pack all weights to f16 fragment layout ----
  for (int l = 0; l < 4; ++l) {
    int b = 5 + 13 * l;
    int h = (l == 0) ? 256 : 128;
    const float* ew1 = (const float*)d_in[b + 7];
    const float* ew2 = (const float*)d_in[b + 8];
    const float* nw1 = (const float*)d_in[b + 11];
    const float* nw2 = (const float*)d_in[b + 12];
    k_pack_b<<<cdiv(288 * h, 256), 256, 0, stream>>>(ew1, pe1[l], 273, 288, h);
    k_pack_b<<<cdiv(h * 16, 256), 256, 0, stream>>>(ew2, pe2[l], h, h, 16);
    k_pack_b<<<cdiv(160 * h, 256), 256, 0, stream>>>(nw1, pn1[l], 147, 160, h);
    k_pack_b<<<cdiv(h * 128, 256), 256, 0, stream>>>(nw2, pn2[l], h, h, 128);
  }
  k_pack_b<<<cdiv(128 * 128, 256), 256, 0, stream>>>(post_w, ppost, 128, 128, 128);

  k_f32_to_f16<<<cdiv(E * 16, 256), 256, 0, stream>>>(in_eattr, eattr16, E * 16);

  const float* nf_cur = in_nf;
  const float* coords_cur = in_coords;
  const int Etiles = E / 16;    // 20000
  const int Ntiles = N / 16;    // 1250

  for (int l = 0; l < 4; ++l) {
    int b = 5 + 13 * l;
    int h = (l == 0) ? 256 : 128;
    const float* cb1 = (const float*)d_in[b + 0];
    const float* cmb = (const float*)d_in[b + 1];
    const float* cmw = (const float*)d_in[b + 2];
    const float* cw1 = (const float*)d_in[b + 3];
    const float* cw2 = (const float*)d_in[b + 4];
    const float* eb1 = (const float*)d_in[b + 5];
    const float* eb2 = (const float*)d_in[b + 6];
    const float* nb1 = (const float*)d_in[b + 9];
    const float* nb2 = (const float*)d_in[b + 10];

    k_f32_to_f16<<<cdiv(N * 128, 256), 256, 0, stream>>>(nf_cur, nf16, N * 128);
    k_radial<<<cdiv(E, 256), 256, 0, stream>>>(coords_cur, src, dst, radial16, normed, E);
    k_copy_f32<<<cdiv(N * 3, 256), 256, 0, stream>>>(coords2, coords_cur, N * 3);
    k_zero_f32<<<cdiv(N * 16, 256), 256, 0, stream>>>(agg, N * 16);

    k_edge_mlp<<<cdiv(Etiles, WPB), 128, 0, stream>>>(
        nf16, eattr16, radial16, src, dst, pe1[l], eb1, pe2[l], eb2, edge_out, Etiles, h);

    k_coord_agg<<<cdiv(E, 256), 256, 0, stream>>>(
        edge_out, normed, src, cw1, cb1, cw2, coords2, agg, eattr16, E);

    float* cnext = (l & 1) ? cB : cA;
    k_node_extra<<<cdiv(N, 256), 256, 0, stream>>>(coords2, cmw, cmb, agg, cnext, extra, N);

    float* nfnext = (l & 1) ? nfB : nfA;
    k_node_mlp<<<cdiv(Ntiles, WPB), 128, 0, stream>>>(
        nf16, extra, pn1[l], nb1, pn2[l], nb2, nfnext, Ntiles, h);

    nf_cur = nfnext;
    coords_cur = cnext;
  }

  float* out_embs   = (float*)d_out;
  float* out_graph  = out_embs + (size_t)N * 128;
  float* out_coords = out_graph + (size_t)NG * 128;

  k_f32_to_f16<<<cdiv(N * 128, 256), 256, 0, stream>>>(nf_cur, nf16, N * 128);
  k_post<<<cdiv(Ntiles * 8, WPB), 128, 0, stream>>>(nf16, ppost, post_b, out_embs, Ntiles);

  k_zero_f32<<<cdiv(NG * 128, 256), 256, 0, stream>>>(sums, NG * 128);
  k_zero_f32<<<1, 256, 0, stream>>>(counts, NG);
  k_pool_sum<<<cdiv(N * 128, 256), 256, 0, stream>>>(out_embs, batch, sums, N);
  k_pool_cnt<<<cdiv(N, 256), 256, 0, stream>>>(batch, counts, N);
  k_pool_fin<<<cdiv(NG * 128, 256), 256, 0, stream>>>(sums, counts, out_graph);
  k_copy_f32<<<cdiv(N * 3, 256), 256, 0, stream>>>(out_coords, coords_cur, N * 3);
}
